// SinkhornDistance_28896539967628
// MI455X (gfx1250) — matrix operand order repaired and checked by reading, not compiled, not run on hardware
//
#include <hip/hip_runtime.h>
#include <math.h>

// ---- problem constants (match reference) ----
constexpr int   BB = 4;
constexpr int   NN = 1024;
constexpr int   MM = 1024;
constexpr int   DD = 32;
constexpr int   K_ITERS = 50;
constexpr float K_EPS   = 0.01f;
constexpr float K_CLIP  = 1e-8f;
constexpr float K_MAXD  = 1000000.0f;
// mu = nu = 1/(1024 + 1e-8)
constexpr float K_MUNU  = 1.0f / (1024.0f + 1e-8f);

typedef __attribute__((ext_vector_type(2)))  float    v2f;
typedef __attribute__((ext_vector_type(8)))  float    v8f;
typedef __attribute__((ext_vector_type(16))) _Float16 v16h;

#if defined(__has_builtin)
#if __has_builtin(__builtin_amdgcn_wmma_f32_16x16x4_f32)
#define USE_F32_WMMA 1
#endif
#endif

// Compute a 16x16 tile of dot[r] = sum_d x[i0+M][d]*y[j0+N][d]
// with, per the CDNA5 C/D layout: M = r + 8*half, N = lane&15.
// xrow points at x[b][i0 + (lane&15)]*DD, yrow at y[b][j0 + (lane&15)]*DD.
__device__ __forceinline__ v8f tile_dot(const float* __restrict__ xrow,
                                        const float* __restrict__ yrow,
                                        int half) {
  v8f acc = {};
#if defined(USE_F32_WMMA)
  // A 16x4 f32 layout: VGPR0 = K (lanes 0-15) / K+2 (lanes 16-31); VGPR1 = K+1 / K+3
#pragma unroll
  for (int k0 = 0; k0 < DD; k0 += 4) {
    int kb = k0 + half * 2;
    v2f a; a.x = xrow[kb]; a.y = xrow[kb + 1];
    v2f b; b.x = yrow[kb]; b.y = yrow[kb + 1];
    acc = __builtin_amdgcn_wmma_f32_16x16x4_f32(
        false, a, false, b, (short)0, acc, false, false);
  }
#else
  // Fallback: single f16 16x16x32 WMMA (probe-verified builtin).
  v16h a, b;
#pragma unroll
  for (int t = 0; t < 8; ++t) {
    a[t]     = (_Float16)xrow[half * 8 + t];        // K = half*8 + t
    a[t + 8] = (_Float16)xrow[16 + half * 8 + t];   // K = 16 + half*8 + t
  }
#pragma unroll
  for (int t = 0; t < 16; ++t)
    b[t] = (_Float16)yrow[(t & 8) * 2 + half * 8 + (t & 7)];  // K = (t<8 ? half*8 + t : 16 + half*8 + t-8)
  acc = __builtin_amdgcn_wmma_f32_16x16x32_f16(
      false, a, false, b, (short)0, acc, false, false);
#endif
  return acc;
}

// ---- kernel 1: row norms ||x||^2, ||y||^2 ----
__global__ void sink_norms(const float* __restrict__ x, const float* __restrict__ y,
                           float* __restrict__ xn2, float* __restrict__ yn2) {
  int i = blockIdx.x * blockDim.x + threadIdx.x;
  const int total = BB * NN;
  const float4* src;
  float* dst;
  if (i < total) {
    src = (const float4*)(x + (size_t)i * DD);
    dst = xn2 + i;
  } else if (i < 2 * total) {
    int j = i - total;
    src = (const float4*)(y + (size_t)j * DD);
    dst = yn2 + j;
  } else {
    return;
  }
  float s = 0.f;
#pragma unroll
  for (int t = 0; t < DD / 4; ++t) {
    float4 q = src[t];
    s += q.x * q.x + q.y * q.y + q.z * q.z + q.w * q.w;
  }
  *dst = s;
}

// ---- kernel 2: K = exp(-C/eps), C = max(|x|^2+|y|^2-2 x.y, clip); one wave per 16x16 tile ----
__global__ void __launch_bounds__(256)
sink_cost_k(const float* __restrict__ x, const float* __restrict__ y,
            const float* __restrict__ xn2, const float* __restrict__ yn2,
            float* __restrict__ Kmat) {
  int gwave = (int)((blockIdx.x * blockDim.x + threadIdx.x) >> 5);
  int lane  = threadIdx.x & 31;
  int tj = gwave & 63;
  int ti = (gwave >> 6) & 63;
  int b  = gwave >> 12;
  int lm = lane & 15, half = lane >> 4;
  int i0 = ti * 16, j0 = tj * 16;
  const float* xrow = x + ((size_t)b * NN + i0 + lm) * DD;
  const float* yrow = y + ((size_t)b * MM + j0 + lm) * DD;
  v8f dot = tile_dot(xrow, yrow, half);
  int col = j0 + lm;
  float yn = yn2[b * MM + col];
#pragma unroll
  for (int r = 0; r < 8; ++r) {
    int row = i0 + r + half * 8;
    float C = fmaxf(xn2[b * NN + row] + yn - 2.0f * dot[r], K_CLIP);
    Kmat[((size_t)b * NN + row) * MM + col] = expf(-C * (1.0f / K_EPS));
  }
}

// ---- kernel 3: init u, v ----
__global__ void sink_init(float* __restrict__ u, float* __restrict__ v) {
  int i = blockIdx.x * blockDim.x + threadIdx.x;
  if (i < BB * NN) u[i] = fmaxf(1.0f / (float)NN, K_CLIP);
  if (i < BB * MM) v[i] = fmaxf(1.0f / (float)MM, K_CLIP);
}

// ---- kernel 4: v update: v[m] = max(exp(-eps*log(max(sum_n u*K, clip)) - nu), clip) ----
// block = 256 threads = 16 columns x 16 n-partitions; grid = B * 64
__global__ void __launch_bounds__(256)
sink_v_update(const float* __restrict__ Kmat, const float* __restrict__ u,
              float* __restrict__ v) {
  __shared__ float red[256];
  int tid = threadIdx.x;
  int c = tid & 15, p = tid >> 4;
  int blk = blockIdx.x;
  int m0 = (blk & 63) * 16;
  int b  = blk >> 6;
  int m  = m0 + c;
  const float* Kb = Kmat + (size_t)b * NN * MM;
  const float* ub = u + b * NN;
  float s = 0.f;
  int nbase = p * (NN / 16);
#pragma unroll 4
  for (int i = 0; i < NN / 16; ++i) {
    int n = nbase + i;
    s += ub[n] * Kb[(size_t)n * MM + m];
  }
  red[tid] = s;
  __syncthreads();
  for (int off = 128; off >= 16; off >>= 1) {
    if (tid < off) red[tid] += red[tid + off];
    __syncthreads();
  }
  if (tid < 16) {
    float su = fmaxf(red[tid], K_CLIP);
    v[b * MM + m0 + tid] = fmaxf(expf(-K_EPS * logf(su) - K_MUNU), K_CLIP);
  }
}

// ---- kernel 5: u update: u[n] = max(exp(-eps*log(max(sum_m v*K, clip)) - mu), clip) ----
// block = 256 threads = 8 rows x 32 lanes; grid = B * 128
__global__ void __launch_bounds__(256)
sink_u_update(const float* __restrict__ Kmat, const float* __restrict__ v,
              float* __restrict__ u) {
  int tid = threadIdx.x;
  int lane = tid & 31, r = tid >> 5;
  int blk = blockIdx.x;
  int n0 = (blk & 127) * 8;
  int b  = blk >> 7;
  int n  = n0 + r;
  const float* Krow = Kmat + ((size_t)b * NN + n) * MM;
  const float* vb = v + b * MM;
  float s = 0.f;
#pragma unroll 4
  for (int i = 0; i < MM / 32; ++i) {
    int m = i * 32 + lane;
    s += vb[m] * Krow[m];
  }
#pragma unroll
  for (int off = 16; off >= 1; off >>= 1) s += __shfl_down(s, off, 32);
  if (lane == 0) {
    float sv = fmaxf(s, K_CLIP);
    u[b * NN + n] = fmaxf(expf(-K_EPS * logf(sv) - K_MUNU), K_CLIP);
  }
}

// ---- kernel 6: per-tile partial of sum(T*C); recompute C via WMMA; one wave per tile ----
__global__ void __launch_bounds__(256)
sink_final_tile(const float* __restrict__ x, const float* __restrict__ y,
                const float* __restrict__ xn2, const float* __restrict__ yn2,
                const float* __restrict__ Kmat, const float* __restrict__ u,
                const float* __restrict__ v, float* __restrict__ partials) {
  int gwave = (int)((blockIdx.x * blockDim.x + threadIdx.x) >> 5);
  int lane  = threadIdx.x & 31;
  int tj = gwave & 63;
  int ti = (gwave >> 6) & 63;
  int b  = gwave >> 12;
  int lm = lane & 15, half = lane >> 4;
  int i0 = ti * 16, j0 = tj * 16;
  const float* xrow = x + ((size_t)b * NN + i0 + lm) * DD;
  const float* yrow = y + ((size_t)b * MM + j0 + lm) * DD;
  v8f dot = tile_dot(xrow, yrow, half);
  int col = j0 + lm;
  float vc = v[b * MM + col];
  float yn = yn2[b * MM + col];
  float acc = 0.f;
#pragma unroll
  for (int r = 0; r < 8; ++r) {
    int row = i0 + r + half * 8;
    float C  = fmaxf(xn2[b * NN + row] + yn - 2.0f * dot[r], K_CLIP);
    float Kv = Kmat[((size_t)b * NN + row) * MM + col];
    acc += u[b * NN + row] * Kv * vc * C;
  }
#pragma unroll
  for (int off = 16; off >= 1; off >>= 1) acc += __shfl_down(acc, off, 32);
  if (lane == 0) partials[b * 4096 + ti * 64 + tj] = acc;
}

// ---- kernel 7: deterministic per-batch reduction + clip + nan_to_num ----
__global__ void sink_final_reduce(const float* __restrict__ partials,
                                  float* __restrict__ out) {
  __shared__ float red[256];
  int b = blockIdx.x, tid = threadIdx.x;
  float s = 0.f;
  for (int i = tid; i < 4096; i += 256) s += partials[b * 4096 + i];
  red[tid] = s;
  __syncthreads();
  for (int off = 128; off >= 1; off >>= 1) {
    if (tid < off) red[tid] += red[tid + off];
    __syncthreads();
  }
  if (tid == 0) {
    float d = red[0];
    if (isnan(d)) d = 0.f;                 // jnp: clip(NaN)=NaN -> nan_to_num -> 0
    d = fminf(fmaxf(d, K_CLIP), K_MAXD);   // clip handles +/-inf like reference
    out[b] = d;
  }
}

extern "C" void kernel_launch(void* const* d_in, const int* in_sizes, int n_in,
                              void* d_out, int out_size, void* d_ws, size_t ws_size,
                              hipStream_t stream) {
  (void)in_sizes; (void)n_in; (void)out_size; (void)ws_size;
  const float* x = (const float*)d_in[0];   // [4,1024,32]
  const float* y = (const float*)d_in[1];   // [4,1024,32]
  float* out = (float*)d_out;               // [4]

  // workspace layout (floats): K | xn2 | yn2 | u | v | partials  (~16.9 MB)
  float* ws       = (float*)d_ws;
  float* Kmat     = ws;                               // 4*1024*1024
  float* xn2      = Kmat + (size_t)BB * NN * MM;      // 4096
  float* yn2      = xn2 + BB * NN;                    // 4096
  float* u        = yn2 + BB * MM;                    // 4096
  float* v        = u + BB * NN;                      // 4096
  float* partials = v + BB * MM;                      // 4*4096

  const int tile_waves  = BB * 64 * 64;               // 16384 tiles
  const int tile_blocks = tile_waves * 32 / 256;      // 2048 blocks

  sink_norms<<<(2 * BB * NN + 255) / 256, 256, 0, stream>>>(x, y, xn2, yn2);
  sink_cost_k<<<tile_blocks, 256, 0, stream>>>(x, y, xn2, yn2, Kmat);
  sink_init<<<(BB * NN + 255) / 256, 256, 0, stream>>>(u, v);

  for (int it = 0; it < K_ITERS; ++it) {
    sink_v_update<<<BB * 64, 256, 0, stream>>>(Kmat, u, v);
    sink_u_update<<<BB * 128, 256, 0, stream>>>(Kmat, v, u);
  }

  sink_final_tile<<<tile_blocks, 256, 0, stream>>>(x, y, xn2, yn2, Kmat, u, v, partials);
  sink_final_reduce<<<BB, 256, 0, stream>>>(partials, out);
}
